// DiagLinear_45698452030176
// MI455X (gfx1250) — matrix operand order repaired and checked by reading, not compiled, not run on hardware
//
#include <hip/hip_runtime.h>

// Problem constants (from the reference)
#define PPART 16
#define DIN   256
#define DOUT  256
#define COLS  4096              // P*DIN == P*DOUT
#define NTOK  16384             // 8 * 2048
#define TOKB  32                // tokens per block
#define NMT   2                 // m-tiles per wave  (2*16 = 32 tokens)
#define NOT   8                 // o-tiles per o-group
#define NOG   2                 // o-groups (8*2*16 = 256 = DOUT)
#define WELEMS (PPART * DOUT * DIN)   // 1,048,576 weight elements

typedef __attribute__((ext_vector_type(16))) __bf16         v16bf;
typedef __attribute__((ext_vector_type(16))) unsigned short v16u;
typedef __attribute__((ext_vector_type(8)))  float          v8f;

// Truncation-based hi/lo split: hi = f with low 16 mantissa bits cleared
// (exactly representable in bf16, widened form is free), lo = RNE(f - hi).
static __device__ __forceinline__ void split16_trunc(const float* fa, v16bf& hi, v16bf& lo) {
  #pragma unroll
  for (int e = 0; e < 16; ++e) {
    const float hf = __builtin_bit_cast(float,
                       __builtin_bit_cast(unsigned, fa[e]) & 0xffff0000u);
    hi[e] = (__bf16)hf;              // exact (no rounding)
    lo[e] = (__bf16)(fa[e] - hf);    // one RNE convert
  }
}

// One-shot weight conversion: fp32 -> bf16 hi/lo bit patterns in workspace.
__global__ void convert_w_kernel(const float* __restrict__ w,
                                 unsigned short* __restrict__ whi,
                                 unsigned short* __restrict__ wlo) {
  for (int e = blockIdx.x * blockDim.x + threadIdx.x; e < WELEMS;
       e += gridDim.x * blockDim.x) {
    const float f  = w[e];
    const unsigned u = __builtin_bit_cast(unsigned, f);
    const float hf = __builtin_bit_cast(float, u & 0xffff0000u);
    whi[e] = (unsigned short)(u >> 16);
    wlo[e] = __builtin_bit_cast(unsigned short, (__bf16)(f - hf));
  }
}

template <bool PRECONV>
__global__ __launch_bounds__(512)
void diag_linear_bf16x3(const float* __restrict__ x,
                        const float* __restrict__ w,
                        const unsigned short* __restrict__ whi,
                        const unsigned short* __restrict__ wlo,
                        const float* __restrict__ bias,
                        float* __restrict__ out)
{
  // 32 KB staging tile for the (p,o) -> o*16+p store transpose
  __shared__ float lds[TOKB * 256];

  const int tid    = threadIdx.x;
  const int p      = tid >> 5;          // wave id == partition (16 waves)
  const int lane   = tid & 31;
  const int l15    = lane & 15;
  const int hiHalf = lane >> 4;         // 0 or 1
  const int t0     = blockIdx.x * TOKB;

  const float* xp = x + (size_t)p * DIN;          // + t*COLS + k
  const size_t wbase = (size_t)p * DOUT * DIN;    // + o*DIN + i

  // A-operand K sub-offset per lane half (contiguous, matching B:
  // lanes 0-15 hold K {0..15}, lanes 16-31 hold K {16..31})
  const int aoff = hiHalf * 16;

  for (int og = 0; og < NOG; ++og) {
    const int obase = og * (NOT * 16);

    // Accumulators initialized with the bias (bias depends only on the
    // output column o*16+p, constant across the 8 C-VGPRs of a lane).
    v8f acc[NMT][NOT];
    #pragma unroll
    for (int ot = 0; ot < NOT; ++ot) {
      const float bv = bias[(obase + ot * 16 + l15) * PPART + p];
      #pragma unroll
      for (int mt = 0; mt < NMT; ++mt)
        #pragma unroll
        for (int r = 0; r < 8; ++r) acc[mt][ot][r] = bv;
    }

    for (int kc = 0; kc < DIN / 32; ++kc) {
      const int kbase = kc * 32;

      // ---- load + hi/lo-split A tiles (x rows, one 16-float run) ----
      v16bf ahi[NMT], alo[NMT];
      #pragma unroll
      for (int mt = 0; mt < NMT; ++mt) {
        const float* ap = xp + (size_t)(t0 + mt * 16 + l15) * COLS + kbase + aoff;
        float4 f0 = *(const float4*)(ap);
        float4 f1 = *(const float4*)(ap + 4);
        float4 f2 = *(const float4*)(ap + 8);
        float4 f3 = *(const float4*)(ap + 12);
        float fa[16] = {f0.x,f0.y,f0.z,f0.w, f1.x,f1.y,f1.z,f1.w,
                        f2.x,f2.y,f2.z,f2.w, f3.x,f3.y,f3.z,f3.w};
        split16_trunc(fa, ahi[mt], alo[mt]);
        __builtin_prefetch(ap + 32, 0, 1);   // next K chunk
      }

      // ---- per o-tile: B operand (weight rows), 3-pass WMMA ----
      #pragma unroll
      for (int ot = 0; ot < NOT; ++ot) {
        // B 32x16: lanes 0-15 hold K {0..15}, lanes 16-31 K {16..31} (contiguous)
        const size_t boff = wbase + (size_t)(obase + ot * 16 + l15) * DIN
                                  + kbase + hiHalf * 16;
        v16bf bhi, blo;
        if (PRECONV) {
          // pre-converted bf16: one 32-byte run each (two global_load_b128)
          bhi = __builtin_bit_cast(v16bf, *(const v16u*)(whi + boff));
          blo = __builtin_bit_cast(v16bf, *(const v16u*)(wlo + boff));
        } else {
          const float* bp = w + boff;
          float4 g0 = *(const float4*)(bp);
          float4 g1 = *(const float4*)(bp + 4);
          float4 g2 = *(const float4*)(bp + 8);
          float4 g3 = *(const float4*)(bp + 12);
          float fb[16] = {g0.x,g0.y,g0.z,g0.w, g1.x,g1.y,g1.z,g1.w,
                          g2.x,g2.y,g2.z,g2.w, g3.x,g3.y,g3.z,g3.w};
          split16_trunc(fb, bhi, blo);
        }

        #pragma unroll
        for (int mt = 0; mt < NMT; ++mt) {
          acc[mt][ot] = __builtin_amdgcn_wmma_f32_16x16x32_bf16(
              false, alo[mt], false, bhi, (short)0, acc[mt][ot], false, false);
          acc[mt][ot] = __builtin_amdgcn_wmma_f32_16x16x32_bf16(
              false, ahi[mt], false, blo, (short)0, acc[mt][ot], false, false);
          acc[mt][ot] = __builtin_amdgcn_wmma_f32_16x16x32_bf16(
              false, ahi[mt], false, bhi, (short)0, acc[mt][ot], false, false);
        }
      }
    }

    // ---- store: transpose (p,o) -> o*16+p through LDS, coalesced out ----
    for (int ot = 0; ot < NOT; ++ot) {
      __syncthreads();                      // previous tile fully consumed
      #pragma unroll
      for (int mt = 0; mt < NMT; ++mt) {
        #pragma unroll
        for (int r = 0; r < 8; ++r) {
          const int trow = mt * 16 + r + hiHalf * 8;   // C layout: +8 rows hi lanes
          lds[trow * 256 + l15 * 16 + p] = acc[mt][ot][r];
        }
      }
      __syncthreads();
      const int cbase = (obase + ot * 16) * PPART;     // contiguous 256-col span
      const float4* l4 = (const float4*)lds;
      #pragma unroll
      for (int rep = 0; rep < (TOKB * 256 / 4) / 512; ++rep) {
        const int linear = rep * 512 + tid;
        const int row = linear >> 6;        // 64 float4 per 256-float row
        const int c4  = linear & 63;
        *(float4*)(out + (size_t)(t0 + row) * COLS + cbase + c4 * 4) = l4[linear];
      }
    }
  }
}

extern "C" void kernel_launch(void* const* d_in, const int* in_sizes, int n_in,
                              void* d_out, int out_size, void* d_ws, size_t ws_size,
                              hipStream_t stream) {
  (void)in_sizes; (void)n_in; (void)out_size;
  const float* x    = (const float*)d_in[0];   // (8,2048,4096) fp32
  const float* w    = (const float*)d_in[1];   // (16,256,256)  fp32
  const float* bias = (const float*)d_in[2];   // (4096,)       fp32
  float* out = (float*)d_out;

  dim3 grid(NTOK / TOKB);            // 512 blocks
  dim3 block(512);                   // 16 waves, wave w <-> partition p=w

  const size_t need = 2ull * WELEMS * sizeof(unsigned short);  // 4 MB
  if (ws_size >= need) {
    unsigned short* whi = (unsigned short*)d_ws;
    unsigned short* wlo = whi + WELEMS;
    convert_w_kernel<<<256, 256, 0, stream>>>(w, whi, wlo);
    diag_linear_bf16x3<true><<<grid, block, 0, stream>>>(x, w, whi, wlo, bias, out);
  } else {
    diag_linear_bf16x3<false><<<grid, block, 0, stream>>>(x, w, nullptr, nullptr, bias, out);
  }
}
